// Mamba2Simple_25108378812677
// MI455X (gfx1250) — compile-verified, matrix operand order checked by reading, not run on hardware
//
#include <hip/hip_runtime.h>
#include <hip/hip_bf16.h>

// ---------------------------------------------------------------------------
// Mamba2 forward for MI455X (gfx1250).
//  - Big GEMMs: bf16x3 split-precision on V_WMMA_F32_16X16X32_BF16.
//    Operands pre-split to bf16 hi/lo in global memory, then streamed into
//    LDS with GLOBAL_LOAD_ASYNC_TO_LDS_B128 (ASYNCcnt-pipelined, double
//    buffered) -> zero conversion VALU and zero VGPR staging in the hot loop.
//  - SSD chunk math: native fp32 V_WMMA_F32_16X16X4_F32.
// ---------------------------------------------------------------------------

typedef __attribute__((ext_vector_type(2)))  float  f32x2;
typedef __attribute__((ext_vector_type(8)))  float  f32x8;
typedef __attribute__((ext_vector_type(16))) __bf16 bf16x16;

#define D_MODEL   2048
#define D_STATE   64
#define D_CONV    4
#define D_HEAD    128
#define NHEADS    32
#define CHUNK     256
#define NCHUNK    16
#define D_INNER   4096
#define CONV_DIM  4224
#define D_IN_PROJ 8352
#define BATCH     2
#define SEQLEN    4096
#define NTOK      (BATCH * SEQLEN)      // 8192
#define XBC_OFF   D_INNER
#define BCOL      (D_INNER)             // 4096
#define CCOL      (D_INNER + D_STATE)   // 4160
#define DT_OFF    (D_IN_PROJ - NHEADS)  // 8320

__device__ __forceinline__ f32x8 zero8() {
  f32x8 z;
  for (int i = 0; i < 8; ++i) z[i] = 0.f;
  return z;
}

__device__ __forceinline__ f32x8 wmma4(f32x2 a, f32x2 b, f32x8 c) {
  return __builtin_amdgcn_wmma_f32_16x16x4_f32(false, a, false, b, (short)0, c,
                                               false, false);
}

__device__ __forceinline__ f32x8 wmma_bf16(bf16x16 a, bf16x16 b, f32x8 c) {
  return __builtin_amdgcn_wmma_f32_16x16x32_bf16(false, a, false, b, (short)0,
                                                 c, false, false);
}

union Frag {
  uint4   q[2];
  bf16x16 v;
};

// split f into bf16 hi + bf16 lo (hi = truncate, lo = bf16(f - hi))
__device__ __forceinline__ void cvt_hilo(float f, unsigned& hi, unsigned& lo) {
  unsigned u = __float_as_uint(f);
  hi = u >> 16;
  float fhi = __uint_as_float(u & 0xffff0000u);
  lo = __float_as_uint(f - fhi) >> 16;
}

__device__ __forceinline__ void pack4(float4 f, uint2& h, uint2& l) {
  unsigned h0, h1, h2, h3, l0, l1, l2, l3;
  cvt_hilo(f.x, h0, l0);
  cvt_hilo(f.y, h1, l1);
  cvt_hilo(f.z, h2, l2);
  cvt_hilo(f.w, h3, l3);
  h.x = h0 | (h1 << 16);  h.y = h2 | (h3 << 16);
  l.x = l0 | (l1 << 16);  l.y = l2 | (l3 << 16);
}

// --- CDNA5 async global->LDS (16B per lane), tracked by ASYNCcnt -----------
__device__ __forceinline__ void async_cp16(unsigned lds_byte_off, const void* g) {
  asm volatile("global_load_async_to_lds_b128 %0, %1, off"
               :: "v"(lds_byte_off), "v"(g)
               : "memory");
}
__device__ __forceinline__ void wait_async_le6() {
  asm volatile("s_wait_asynccnt 0x6" ::: "memory");
}
__device__ __forceinline__ void wait_async_0() {
  asm volatile("s_wait_asynccnt 0x0" ::: "memory");
}

// ---------------------------------------------------------------------------
// Elementwise fp32 -> (bf16 hi, bf16 lo), 4 elements per thread.
// ---------------------------------------------------------------------------
__global__ void split_bf16(const float* __restrict__ X,
                           unsigned short* __restrict__ H,
                           unsigned short* __restrict__ L, long n4) {
  long i = (long)blockIdx.x * 256 + threadIdx.x;
  if (i >= n4) return;
  float4 f = ((const float4*)X)[i];
  uint2 h, l;
  pack4(f, h, l);
  ((uint2*)H)[i] = h;
  ((uint2*)L)[i] = l;
}

// ---------------------------------------------------------------------------
// GEMM: C[M,N] = A[M,K] * Bt[N,K]^T, bf16x3, operands pre-split (hi/lo).
// Block tile 128x64x32, 8 waves (4Mx2N), wave tile 32x32.
// LDS rows padded to 40 bf16 (80B): conflict-free, 16B-aligned b128 reads.
// Staging: per-thread 6 x GLOBAL_LOAD_ASYNC_TO_LDS_B128 per K-tile,
// double-buffered on ASYNCcnt.
// ---------------------------------------------------------------------------
#define BM 128
#define BN 64
#define BK 32
#define LP 20                 // dwords per LDS row
#define ABUF (BM * LP * 4)    // bytes per A buffer
#define BBUF (BN * LP * 4)    // bytes per B buffer

__global__ __launch_bounds__(256) void gemm_abt_bf16x3(
    const unsigned short* __restrict__ Ah, const unsigned short* __restrict__ Al,
    const unsigned short* __restrict__ Bh, const unsigned short* __restrict__ Bl,
    float* __restrict__ C, int M, int N, int K) {
  __shared__ unsigned AsH[2][BM * LP];
  __shared__ unsigned AsL[2][BM * LP];
  __shared__ unsigned BsH[2][BN * LP];
  __shared__ unsigned BsL[2][BN * LP];

  const int tid  = threadIdx.x;
  const int wid  = tid >> 5;
  const int lane = tid & 31;
  const int half = lane >> 4;
  const int l16  = lane & 15;
  const int bm = blockIdx.y * BM;
  const int bn = blockIdx.x * BN;
  const int waveM = wid >> 1;  // 0..3
  const int waveN = wid & 1;   // 0..1

  const unsigned asH0 = (unsigned)(size_t)&AsH[0][0];
  const unsigned asL0 = (unsigned)(size_t)&AsL[0][0];
  const unsigned bsH0 = (unsigned)(size_t)&BsH[0][0];
  const unsigned bsL0 = (unsigned)(size_t)&BsL[0][0];

  f32x8 acc[2][2];
  for (int i = 0; i < 2; ++i)
    for (int j = 0; j < 2; ++j) acc[i][j] = zero8();

  // 6 async transfers per thread per tile (A hi/lo: 4, B hi/lo: 2).
  // B rows are clamped (not branched) so the per-wave ASYNCcnt is always 6;
  // clamped garbage only reaches output columns >= N, which are never stored.
  auto issue_async = [&](int buf, int k0) {
    for (int it = 0; it < 2; ++it) {
      int idx = it * 256 + tid;          // 0..511
      int row = idx >> 2;                // 0..127
      int ch  = idx & 3;                 // 16B chunk in 64B row
      unsigned ldso = (unsigned)(row * 80 + ch * 16);
      size_t go = (size_t)(bm + row) * K + k0 + ch * 8;
      async_cp16(asH0 + buf * ABUF + ldso, Ah + go);
      async_cp16(asL0 + buf * ABUF + ldso, Al + go);
    }
    {
      int row = tid >> 2;                // 0..63
      int ch  = tid & 3;
      int rc  = (bn + row < N) ? (bn + row) : (N - 1);
      unsigned ldso = (unsigned)(row * 80 + ch * 16);
      size_t go = (size_t)rc * K + k0 + ch * 8;
      async_cp16(bsH0 + buf * BBUF + ldso, Bh + go);
      async_cp16(bsL0 + buf * BBUF + ldso, Bl + go);
    }
  };

  auto compute = [&](int buf) {
    Frag aH[2], aL[2], bH[2], bL[2];
    for (int i = 0; i < 2; ++i) {
      int r = (waveM * 32 + i * 16 + l16) * LP;
      aH[i].q[0] = *(const uint4*)&AsH[buf][r + half * 4];
      aH[i].q[1] = *(const uint4*)&AsH[buf][r + 8 + half * 4];
      aL[i].q[0] = *(const uint4*)&AsL[buf][r + half * 4];
      aL[i].q[1] = *(const uint4*)&AsL[buf][r + 8 + half * 4];
    }
    for (int j = 0; j < 2; ++j) {
      int r = (waveN * 32 + j * 16 + l16) * LP;
      bH[j].q[0] = *(const uint4*)&BsH[buf][r + half * 4];
      bH[j].q[1] = *(const uint4*)&BsH[buf][r + 8 + half * 4];
      bL[j].q[0] = *(const uint4*)&BsL[buf][r + half * 4];
      bL[j].q[1] = *(const uint4*)&BsL[buf][r + 8 + half * 4];
    }
    for (int i = 0; i < 2; ++i)
      for (int j = 0; j < 2; ++j) {
        acc[i][j] = wmma_bf16(aH[i].v, bH[j].v, acc[i][j]);
        acc[i][j] = wmma_bf16(aL[i].v, bH[j].v, acc[i][j]);
        acc[i][j] = wmma_bf16(aH[i].v, bL[j].v, acc[i][j]);
      }
  };

  const int NT = K / BK;
  issue_async(0, 0);
  for (int kt = 0; kt < NT; ++kt) {
    if (kt + 1 < NT) {
      issue_async((kt + 1) & 1, (kt + 1) * BK);  // overlap next tile
      wait_async_le6();                          // FIFO: current tile done
    } else {
      wait_async_0();
    }
    __syncthreads();          // all waves' transfers for this buffer visible
    compute(kt & 1);
    __syncthreads();          // all waves done reading before overwrite
  }

  for (int i = 0; i < 2; ++i)
    for (int j = 0; j < 2; ++j) {
      int col = bn + waveN * 32 + j * 16 + l16;
      if (col < N) {
        for (int e = 0; e < 8; ++e) {
          int row = bm + waveM * 32 + i * 16 + half * 8 + e;
          C[(size_t)row * N + col] = acc[i][j][e];
        }
      }
    }
}

// ---------------------------------------------------------------------------
__global__ void conv_silu(const float* __restrict__ zxbcdt,
                          const float* __restrict__ conv_w,
                          const float* __restrict__ conv_b,
                          float* __restrict__ xbc) {
  long idx = (long)blockIdx.x * 256 + threadIdx.x;
  if (idx >= (long)NTOK * CONV_DIM) return;
  int  ch = (int)(idx % CONV_DIM);
  long t  = idx / CONV_DIM;
  int  l  = (int)(t % SEQLEN);
  long b  = t / SEQLEN;
  float a = conv_b[ch];
  const float* base = zxbcdt + (b * SEQLEN) * (size_t)D_IN_PROJ + XBC_OFF + ch;
  for (int k = 0; k < D_CONV; ++k) {
    int ls = l - (D_CONV - 1) + k;
    if (ls >= 0) a += base[(size_t)ls * D_IN_PROJ] * conv_w[ch * D_CONV + k];
  }
  xbc[(size_t)t * CONV_DIM + ch] = a / (1.f + __expf(-a));
}

__global__ void dt_softplus(const float* __restrict__ zxbcdt,
                            const float* __restrict__ dt_bias,
                            float* __restrict__ dtw) {
  long idx = (long)blockIdx.x * 256 + threadIdx.x;
  if (idx >= (long)NTOK * NHEADS) return;
  int  h = (int)(idx % NHEADS);
  long t = idx / NHEADS;
  float x = zxbcdt[(size_t)t * D_IN_PROJ + DT_OFF + h] + dt_bias[h];
  dtw[idx] = (x > 20.f) ? x : log1pf(__expf(x));
}

// ---------------------------------------------------------------------------
// Per (b, chunk, head): cumsum(dA), Y_diag = (C B^T ⊙ L) @ xdt (fp32 WMMA,
// LDS round-trip between the two matmuls), chunk states = xdt_scaled^T @ B.
// ---------------------------------------------------------------------------
__global__ __launch_bounds__(256) void chunk_fwd(
    const float* __restrict__ xbc, const float* __restrict__ dtw,
    const float* __restrict__ A_log, float* __restrict__ cum_ws,
    float* __restrict__ decay_ws, float* __restrict__ states_ws,
    float* __restrict__ y_ws) {
  const int h = blockIdx.x, c = blockIdx.y, b = blockIdx.z;
  const int tid = threadIdx.x;
  const int wid = tid >> 5, lane = tid & 31, half = lane >> 4, l16 = lane & 15;
  __shared__ float sh_cum[CHUNK];
  __shared__ float sh_dt[CHUNK];
  __shared__ float sh_sc[CHUNK];
  __shared__ float Msh[8][16 * 18];

  const float Aval = -__expf(A_log[h]);
  const size_t tBase = (size_t)b * SEQLEN + (size_t)c * CHUNK;
  const float dtv = dtw[(tBase + tid) * NHEADS + h];
  const float dA = dtv * Aval;

  sh_cum[tid] = dA;
  __syncthreads();
  float v = dA;
  for (int off = 1; off < CHUNK; off <<= 1) {
    float add = (tid >= off) ? sh_cum[tid - off] : 0.f;
    __syncthreads();
    v += add;
    sh_cum[tid] = v;
    __syncthreads();
  }
  const float cumlast = sh_cum[CHUNK - 1];
  sh_dt[tid] = dtv;
  sh_sc[tid] = dtv * __expf(cumlast - v);
  const size_t bch = ((size_t)(b * NCHUNK + c) * NHEADS + h);
  cum_ws[bch * CHUNK + tid] = v;
  if (tid == 0) decay_ws[bch] = __expf(cumlast);
  __syncthreads();

  const float* xrow = xbc + tBase * CONV_DIM;

  for (int T = wid; T < 128; T += 8) {
    const int si = T >> 3, pj = T & 7;
    f32x8 acc = zero8();
    for (int zk = 0; zk <= si; ++zk) {
      f32x8 g = zero8();
      for (int nk = 0; nk < 16; ++nk) {
        f32x2 a  = *(const f32x2*)(xrow + (size_t)(si * 16 + l16) * CONV_DIM +
                                   CCOL + nk * 4 + half * 2);
        f32x2 bb = *(const f32x2*)(xrow + (size_t)(zk * 16 + l16) * CONV_DIM +
                                   BCOL + nk * 4 + half * 2);
        g = wmma4(a, bb, g);
      }
      for (int e = 0; e < 8; ++e) {
        int srow = si * 16 + half * 8 + e;
        int zcol = zk * 16 + l16;
        float f = (zcol <= srow) ? __expf(sh_cum[srow] - sh_cum[zcol]) : 0.f;
        Msh[wid][(half * 8 + e) * 18 + l16] = g[e] * f;
      }
      for (int kk = 0; kk < 16; kk += 4) {
        f32x2 a2 = *(const f32x2*)&Msh[wid][l16 * 18 + kk + half * 2];
        int z0 = zk * 16 + kk + half * 2;
        float x0 = xrow[(size_t)z0 * CONV_DIM + h * D_HEAD + pj * 16 + l16];
        float x1 = xrow[(size_t)(z0 + 1) * CONV_DIM + h * D_HEAD + pj * 16 + l16];
        f32x2 b2;
        b2.x = x0 * sh_dt[z0];
        b2.y = x1 * sh_dt[z0 + 1];
        acc = wmma4(a2, b2, acc);
      }
    }
    for (int e = 0; e < 8; ++e) {
      size_t row = tBase + si * 16 + half * 8 + e;
      y_ws[row * D_INNER + h * D_HEAD + pj * 16 + l16] = acc[e];
    }
  }

  for (int T2 = wid; T2 < 32; T2 += 8) {
    const int pi = T2 >> 2, nj = T2 & 3;
    f32x8 acc = zero8();
    for (int z = 0; z < CHUNK; z += 4) {
      int z0 = z + half * 2;
      f32x2 a;
      a.x = xrow[(size_t)z0 * CONV_DIM + h * D_HEAD + pi * 16 + l16] * sh_sc[z0];
      a.y = xrow[(size_t)(z0 + 1) * CONV_DIM + h * D_HEAD + pi * 16 + l16] * sh_sc[z0 + 1];
      f32x2 bb;
      bb.x = xrow[(size_t)z0 * CONV_DIM + BCOL + nj * 16 + l16];
      bb.y = xrow[(size_t)(z0 + 1) * CONV_DIM + BCOL + nj * 16 + l16];
      acc = wmma4(a, bb, acc);
    }
    for (int e = 0; e < 8; ++e) {
      int p = pi * 16 + half * 8 + e;
      states_ws[bch * (D_HEAD * D_STATE) + (size_t)p * D_STATE + nj * 16 + l16] = acc[e];
    }
  }
}

// ---------------------------------------------------------------------------
__global__ __launch_bounds__(256) void chunk_scan(
    const float* __restrict__ states_ws, const float* __restrict__ decay_ws,
    float* __restrict__ prev_ws) {
  const int bh = blockIdx.x;
  const int b = bh >> 5, h = bh & 31;
  const int tid = threadIdx.x;
  float carry[32];
  for (int e = 0; e < 32; ++e) carry[e] = 0.f;
  for (int c = 0; c < NCHUNK; ++c) {
    size_t bch = ((size_t)(b * NCHUNK + c) * NHEADS + h);
    float dec = decay_ws[bch];
    const float* s = states_ws + bch * (D_HEAD * D_STATE) + (size_t)tid * 32;
    float* pv = prev_ws + bch * (D_HEAD * D_STATE) + (size_t)tid * 32;
    for (int e = 0; e < 32; ++e) {
      pv[e] = carry[e];
      carry[e] = carry[e] * dec + s[e];
    }
  }
}

// ---------------------------------------------------------------------------
__global__ __launch_bounds__(256) void yoff_add(
    const float* __restrict__ xbc, const float* __restrict__ cum_ws,
    const float* __restrict__ prev_ws, const float* __restrict__ Dv,
    float* __restrict__ y_ws) {
  const int h = blockIdx.x, c = blockIdx.y, b = blockIdx.z;
  const int tid = threadIdx.x;
  const int wid = tid >> 5, lane = tid & 31, half = lane >> 4, l16 = lane & 15;
  __shared__ float sh_cum[CHUNK];
  const size_t bch = ((size_t)(b * NCHUNK + c) * NHEADS + h);
  sh_cum[tid] = cum_ws[bch * CHUNK + tid];
  __syncthreads();
  const size_t tBase = (size_t)b * SEQLEN + (size_t)c * CHUNK;
  const float* xrow = xbc + tBase * CONV_DIM;
  const float* ps = prev_ws + bch * (D_HEAD * D_STATE);  // [p][n]
  const float Dh = Dv[h];

  for (int T = wid; T < 128; T += 8) {
    const int si = T >> 3, pj = T & 7;
    f32x8 acc = zero8();
    for (int kk = 0; kk < D_STATE; kk += 4) {
      f32x2 a  = *(const f32x2*)(xrow + (size_t)(si * 16 + l16) * CONV_DIM +
                                 CCOL + kk + half * 2);
      f32x2 bb = *(const f32x2*)(ps + (size_t)(pj * 16 + l16) * D_STATE + kk + half * 2);
      acc = wmma4(a, bb, acc);
    }
    for (int e = 0; e < 8; ++e) {
      int srow = si * 16 + half * 8 + e;
      size_t row = tBase + srow;
      int col = pj * 16 + l16;
      float xv = xrow[(size_t)srow * CONV_DIM + h * D_HEAD + col];
      size_t o = row * D_INNER + h * D_HEAD + col;
      y_ws[o] = y_ws[o] + acc[e] * __expf(sh_cum[srow]) + xv * Dh;
    }
  }
}

// ---------------------------------------------------------------------------
// gate + RMSNorm; emits yn directly as bf16 hi/lo for the out_proj GEMM.
// ---------------------------------------------------------------------------
__global__ __launch_bounds__(256) void gate_norm(
    const float* __restrict__ y_ws, const float* __restrict__ zxbcdt,
    const float* __restrict__ norm_w,
    unsigned short* __restrict__ ynH, unsigned short* __restrict__ ynL) {
  const int row = blockIdx.x;
  const int tid = threadIdx.x;
  __shared__ float red[256];
  float yg[16];
  float ss = 0.f;
  const float* yr = y_ws + (size_t)row * D_INNER;
  const float* zr = zxbcdt + (size_t)row * D_IN_PROJ;
  for (int i = 0; i < 16; ++i) {
    int cc = tid + i * 256;
    float z = zr[cc];
    float g = z / (1.f + __expf(-z));
    float vv = yr[cc] * g;
    yg[i] = vv;
    ss += vv * vv;
  }
  red[tid] = ss;
  __syncthreads();
  for (int off = 128; off > 0; off >>= 1) {
    if (tid < off) red[tid] += red[tid + off];
    __syncthreads();
  }
  float rs = rsqrtf(red[0] / (float)D_INNER + 1e-5f);
  for (int i = 0; i < 16; ++i) {
    int cc = tid + i * 256;
    float val = yg[i] * rs * norm_w[cc];
    unsigned h, l;
    cvt_hilo(val, h, l);
    ynH[(size_t)row * D_INNER + cc] = (unsigned short)h;
    ynL[(size_t)row * D_INNER + cc] = (unsigned short)l;
  }
}

// ---------------------------------------------------------------------------
extern "C" void kernel_launch(void* const* d_in, const int* in_sizes, int n_in,
                              void* d_out, int out_size, void* d_ws,
                              size_t ws_size, hipStream_t stream) {
  const float* u       = (const float*)d_in[0];
  const float* W_in    = (const float*)d_in[1];
  const float* conv_w  = (const float*)d_in[2];
  const float* conv_b  = (const float*)d_in[3];
  const float* dt_bias = (const float*)d_in[4];
  const float* A_log   = (const float*)d_in[5];
  const float* Dv      = (const float*)d_in[6];
  const float* norm_w  = (const float*)d_in[7];
  const float* W_out   = (const float*)d_in[8];
  float* out = (float*)d_out;

  char* wsb = (char*)d_ws;
  size_t off = 0;
  auto alloc = [&](size_t bytes) {
    void* p = wsb + off;
    off += (bytes + 255) & ~(size_t)255;
    return p;
  };
  float* zxbcdt = (float*)alloc((size_t)NTOK * D_IN_PROJ * 4);
  float* xbc    = (float*)alloc((size_t)NTOK * CONV_DIM * 4);
  float* dtw    = (float*)alloc((size_t)NTOK * NHEADS * 4);
  float* cum    = (float*)alloc((size_t)BATCH * NCHUNK * NHEADS * CHUNK * 4);
  float* decay  = (float*)alloc((size_t)BATCH * NCHUNK * NHEADS * 4);
  float* states = (float*)alloc((size_t)BATCH * NCHUNK * NHEADS * D_HEAD * D_STATE * 4);
  float* prev   = (float*)alloc((size_t)BATCH * NCHUNK * NHEADS * D_HEAD * D_STATE * 4);
  float* y      = (float*)alloc((size_t)NTOK * D_INNER * 4);
  unsigned short* uH  = (unsigned short*)alloc((size_t)NTOK * D_MODEL * 2);
  unsigned short* uL  = (unsigned short*)alloc((size_t)NTOK * D_MODEL * 2);
  unsigned short* wiH = (unsigned short*)alloc((size_t)D_IN_PROJ * D_MODEL * 2);
  unsigned short* wiL = (unsigned short*)alloc((size_t)D_IN_PROJ * D_MODEL * 2);
  unsigned short* woH = (unsigned short*)alloc((size_t)D_MODEL * D_INNER * 2);
  unsigned short* woL = (unsigned short*)alloc((size_t)D_MODEL * D_INNER * 2);
  unsigned short* ynH = (unsigned short*)alloc((size_t)NTOK * D_INNER * 2);
  unsigned short* ynL = (unsigned short*)alloc((size_t)NTOK * D_INNER * 2);

  // 0) pre-split fp32 operands into bf16 hi/lo
  long n4u = (long)NTOK * D_MODEL / 4;
  long n4wi = (long)D_IN_PROJ * D_MODEL / 4;
  long n4wo = (long)D_MODEL * D_INNER / 4;
  split_bf16<<<(int)((n4u + 255) / 256), 256, 0, stream>>>(u, uH, uL, n4u);
  split_bf16<<<(int)((n4wi + 255) / 256), 256, 0, stream>>>(W_in, wiH, wiL, n4wi);
  split_bf16<<<(int)((n4wo + 255) / 256), 256, 0, stream>>>(W_out, woH, woL, n4wo);

  // 1) in_proj GEMM (bf16x3, async LDS): zxbcdt = u @ W_in^T
  gemm_abt_bf16x3<<<dim3((D_IN_PROJ + BN - 1) / BN, NTOK / BM), 256, 0, stream>>>(
      uH, uL, wiH, wiL, zxbcdt, NTOK, D_IN_PROJ, D_MODEL);

  // 2) conv + SiLU, dt softplus
  conv_silu<<<(int)(((long)NTOK * CONV_DIM + 255) / 256), 256, 0, stream>>>(
      zxbcdt, conv_w, conv_b, xbc);
  dt_softplus<<<(int)(((long)NTOK * NHEADS + 255) / 256), 256, 0, stream>>>(
      zxbcdt, dt_bias, dtw);

  // 3) per-chunk: cumsum, Y_diag, states (fp32 WMMA)
  chunk_fwd<<<dim3(NHEADS, NCHUNK, BATCH), 256, 0, stream>>>(
      xbc, dtw, A_log, cum, decay, states, y);

  // 4) inter-chunk sequential scan
  chunk_scan<<<BATCH * NHEADS, 256, 0, stream>>>(states, decay, prev);

  // 5) Y_off + D skip (fp32 WMMA)
  yoff_add<<<dim3(NHEADS, NCHUNK, BATCH), 256, 0, stream>>>(xbc, cum, prev, Dv, y);

  // 6) gate + RMSNorm (emits bf16 hi/lo yn)
  gate_norm<<<NTOK, 256, 0, stream>>>(y, zxbcdt, norm_w, ynH, ynL);

  // 7) out_proj GEMM (bf16x3, async LDS): out = yn @ W_out^T
  gemm_abt_bf16x3<<<dim3(D_MODEL / BN, NTOK / BM), 256, 0, stream>>>(
      ynH, ynL, woH, woL, out, NTOK, D_MODEL, D_INNER);
}
